// DistributedRealSHT_73641509257359
// MI455X (gfx1250) — matrix operand order repaired and checked by reading, not compiled, not run on hardware
//
#include <hip/hip_runtime.h>
#include <stdint.h>

// DistributedRealSHT for MI455X (gfx1250)
// Preferred path (ws >= ~203MB): split-bf16 GEMMs with v_wmma_f32_16x16x32_bf16
//   (x = hi + lo in bf16; D ~= Ah*Bh + Ah*Bl + Al*Bh, f32 accumulate)
//   Stage-2 A tiles staged to LDS via TDM (tensor_load_to_lds, TENSORcnt) for
//   the hi plane and global_load_async_to_lds_b128 (ASYNCcnt) for the lo plane.
// Fallback path: f32 GEMMs with v_wmma_f32_16x16x4_f32.

#define NLAT  256
#define NLON  512
#define LMAX  256
#define MMAX  257
#define MPAD  272                 // 17*16 zero-padded m rows -> unguarded B loads
#define BATCH 4
#define CHANS 64
#define BC    (BATCH * CHANS)     // 256
#define ROWS  (BC * NLAT)         // 65536 DFT rows (b,c,lat)
#define PI_F  3.14159265358979323846f

typedef float  v2f  __attribute__((ext_vector_type(2)));
typedef float  v8f  __attribute__((ext_vector_type(8)));
typedef float  f4v  __attribute__((ext_vector_type(4)));
typedef __bf16 bf16_t;
typedef bf16_t v16bf __attribute__((ext_vector_type(16)));
typedef bf16_t bf8v  __attribute__((ext_vector_type(8)));
typedef bf16_t bf4v  __attribute__((ext_vector_type(4)));
typedef unsigned int u32x4 __attribute__((ext_vector_type(4)));
typedef int          i32x4 __attribute__((ext_vector_type(4)));
typedef int          i32x8 __attribute__((ext_vector_type(8)));

#if defined(__has_builtin)
#  if __has_builtin(__builtin_amdgcn_tensor_load_to_lds)
#    define HAVE_TDM 1
#  endif
#  if __has_builtin(__builtin_amdgcn_s_wait_tensorcnt)
#    define HAVE_TCNT 1
#  endif
#endif

// ---------------- bf16-path ws layout (byte offsets) ----------------
#define F_ELEMS  ((size_t)MPAD * NLON)           // per part: 139264
#define W_ELEMS  ((size_t)MMAX * LMAX * NLAT)    // 16842752
#define Y_ELEMS  ((size_t)2 * MMAX * ROWS)       // 33685504
#define FHI_OFF  ((size_t)0)
#define FLO_OFF  (FHI_OFF + 2 * F_ELEMS * 2)
#define WHI_OFF  (FLO_OFF + 2 * F_ELEMS * 2)
#define WLO_OFF  (WHI_OFF + W_ELEMS * 2)
#define YHI_OFF  (WLO_OFF + W_ELEMS * 2)
#define YLO_OFF  (YHI_OFF + Y_ELEMS * 2)
#define WS_NEED_BF16 (YLO_OFF + Y_ELEMS * 2)     // ~203.2 MB

// ---------------- f32-path ws layout (float offsets) ----------------
#define YOFF_F32  (2 * MPAD * NLON)
#define Y_PART    ((size_t)MMAX * ROWS)

// DFT coefficient: rfft 'forward' norm (1/N) times 2*pi
__device__ __forceinline__ float dft_val(int part, int m, int n) {
  float theta = (2.0f * PI_F / (float)NLON) * (float)((n * m) % NLON);
  float s, c;
  sincosf(theta, &s, &c);
  float scale = 2.0f * PI_F / (float)NLON;
  return (part == 0) ? scale * c : -scale * s;
}

__device__ __forceinline__ void split_bf16(float v, bf16_t& h, bf16_t& l) {
  h = (bf16_t)v;
  l = (bf16_t)(v - (float)h);
}

// assemble a 16-element bf16 fragment from two contiguous 16B pieces
__device__ __forceinline__ v16bf frag2(const bf16_t* p0, const bf16_t* p1) {
  v16bf r;
  *(bf8v*)&r       = *(const bf8v*)p0;
  *((bf8v*)&r + 1) = *(const bf8v*)p1;
  return r;
}

#define WMMA_BF16(A, B, C)                                                   \
  __builtin_amdgcn_wmma_f32_16x16x32_bf16(false, (A), false, (B), (short)0,  \
                                          (C), false, false)
#define WMMA_F32(A, B, C)                                                    \
  __builtin_amdgcn_wmma_f32_16x16x4_f32(false, (A), false, (B), (short)0,    \
                                        (C), false, false)

// per-lane async Global->LDS 16B copy (ASYNCcnt)
__device__ __forceinline__ void async_lds_b128(unsigned lds_off,
                                               unsigned long long gaddr) {
  asm volatile("global_load_async_to_lds_b128 %0, %1, off"
               :: "v"(lds_off), "v"(gaddr) : "memory");
}

// ==================================================================== bf16
// stage 0a: DFT matrix, split into bf16 hi/lo, m-major n-contiguous, padded
__global__ void sht_init_f_bf16(char* __restrict__ wsb) {
  bf16_t* Fhi = (bf16_t*)(wsb + FHI_OFF);
  bf16_t* Flo = (bf16_t*)(wsb + FLO_OFF);
  int idx = blockIdx.x * blockDim.x + threadIdx.x;
  int total = 2 * MPAD * NLON;
  if (idx >= total) return;
  int part = idx / (MPAD * NLON);
  int rem  = idx - part * (MPAD * NLON);
  int m = rem / NLON, n = rem - (rem / NLON) * NLON;
  float v = (m < MMAX) ? dft_val(part, m, n) : 0.0f;
  bf16_t h, l; split_bf16(v, h, l);
  Fhi[idx] = h; Flo[idx] = l;
}

// stage 0b: split weights[m][l][k] into bf16 hi/lo (k stays contiguous)
__global__ void sht_cvt_w(const float* __restrict__ w, char* __restrict__ wsb) {
  bf16_t* Whi = (bf16_t*)(wsb + WHI_OFF);
  bf16_t* Wlo = (bf16_t*)(wsb + WLO_OFF);
  size_t idx = (size_t)blockIdx.x * blockDim.x + threadIdx.x;
  if (idx >= W_ELEMS) return;
  bf16_t h, l; split_bf16(w[idx], h, l);
  Whi[idx] = h; Wlo[idx] = l;
}

// stage 1: DFT-GEMM (65536 x 512 x 2*257), A split during LDS staging
__global__ __launch_bounds__(256)
void sht_dft_bf16(const float* __restrict__ x, char* __restrict__ wsb) {
  __shared__ bf16_t lAhi[16 * NLON];   // 16 KB
  __shared__ bf16_t lAlo[16 * NLON];   // 16 KB
  const bf16_t* Fhi = (const bf16_t*)(wsb + FHI_OFF);
  const bf16_t* Flo = (const bf16_t*)(wsb + FLO_OFF);
  bf16_t* Yhi = (bf16_t*)(wsb + YHI_OFF);
  bf16_t* Ylo = (bf16_t*)(wsb + YLO_OFF);

  const int p0 = blockIdx.x * 16;
  const float* src = x + (size_t)p0 * NLON;    // 16 contiguous rows
  for (int i = threadIdx.x; i < 16 * NLON / 4; i += blockDim.x) {
    f4v v = ((const f4v*)src)[i];
    bf4v h, l;
#pragma unroll
    for (int j = 0; j < 4; ++j) { bf16_t hh, ll; split_bf16(v[j], hh, ll); h[j] = hh; l[j] = ll; }
    ((bf4v*)lAhi)[i] = h;
    ((bf4v*)lAlo)[i] = l;
  }
  __syncthreads();

  const int wave = threadIdx.x >> 5, lane = threadIdx.x & 31;
  const int hl = lane >> 4, ln = lane & 15;

  for (int t = wave; t < 34; t += 8) {           // (part, m_tile) tasks
    const int part = (t >= 17) ? 1 : 0;
    const int m0 = (t - part * 17) * 16;
    const bf16_t* Bh = Fhi + (size_t)part * F_ELEMS + (size_t)(m0 + ln) * NLON;
    const bf16_t* Bl = Flo + (size_t)part * F_ELEMS + (size_t)(m0 + ln) * NLON;
    v8f acc = {};
#pragma unroll 4
    for (int k0 = 0; k0 < NLON; k0 += 32) {
      const int ka = k0 + 8 * hl;                // A: K[ka..ka+8) u K[ka+16..)
      const int kb = k0 + 8 * hl;                // B: K[kb..kb+8) u K[kb+16..)
      v16bf ah = frag2(&lAhi[ln * NLON + ka], &lAhi[ln * NLON + ka + 16]);
      v16bf al = frag2(&lAlo[ln * NLON + ka], &lAlo[ln * NLON + ka + 16]);
      v16bf bh = frag2(&Bh[kb], &Bh[kb + 16]);
      v16bf bl = frag2(&Bl[kb], &Bl[kb + 16]);
      acc = WMMA_BF16(ah, bh, acc);
      acc = WMMA_BF16(ah, bl, acc);
      acc = WMMA_BF16(al, bh, acc);
    }
    const int m = m0 + ln;
    if (m < MMAX) {                              // store guard only, after WMMA
      size_t base = (size_t)part * ((size_t)MMAX * ROWS) + (size_t)m * ROWS + p0 + 8 * hl;
#pragma unroll
      for (int i = 0; i < 8; ++i) {
        bf16_t h, l; split_bf16(acc[i], h, l);   // Y kept as hi/lo pair
        Yhi[base + i] = h; Ylo[base + i] = l;
      }
    }
  }
}

// stage 2: per-m Legendre GEMM; A tiles DMA'd to LDS (TDM + async-LDS)
__global__ __launch_bounds__(256)
void sht_leg_bf16(const char* __restrict__ wsb, float* __restrict__ out) {
  __shared__ bf16_t lAhi[16 * NLAT];   // 8 KB
  __shared__ bf16_t lAlo[16 * NLAT];   // 8 KB
  const bf16_t* Whi = (const bf16_t*)(wsb + WHI_OFF);
  const bf16_t* Wlo = (const bf16_t*)(wsb + WLO_OFF);
  const bf16_t* Yhi = (const bf16_t*)(wsb + YHI_OFF);
  const bf16_t* Ylo = (const bf16_t*)(wsb + YLO_OFF);

  const int m = blockIdx.x, r0 = blockIdx.y * 16, part = blockIdx.z;
  const size_t ybase = (size_t)part * ((size_t)MMAX * ROWS) + (size_t)m * ROWS + (size_t)r0 * NLAT;
  const bf16_t* srch = Yhi + ybase;              // contiguous 16x256 bf16 = 8 KB
  const bf16_t* srcl = Ylo + ybase;

#if defined(HAVE_TDM)
  // hi plane: Tensor Data Mover, 2-D tile 16 rows x 256 bf16, stride 256 elems
  if (threadIdx.x < 32) {
    unsigned           lds_off = (unsigned)(uintptr_t)&lAhi[0];
    unsigned long long ga      = (unsigned long long)(uintptr_t)srch;
    u32x4 g0;
    g0[0] = 1u;                                        // count=1 user D#
    g0[1] = lds_off;                                   // lds_addr (bytes)
    g0[2] = (unsigned)(ga & 0xFFFFFFFFu);              // global_addr[31:0]
    g0[3] = (unsigned)((ga >> 32) & 0x01FFFFFFu) | 0x80000000u; // [56:32]|type=2
    i32x8 g1;
    g1[0] = (int)(1u << 16);        // workgroup_mask=0, data_size=1 (2 bytes)
    g1[1] = (int)(256u << 16);      // tensor_dim0[15:0]=256 in bits 63:48
    g1[2] = (int)(16u << 16);       // tensor_dim0 hi=0 | tensor_dim1 lo=16
    g1[3] = (int)(256u << 16);      // tensor_dim1 hi=0 | tile_dim0=256
    g1[4] = 16;                     // tile_dim1=16, tile_dim2=0
    g1[5] = 256;                    // tensor_dim0_stride lo32 = 256 elems
    g1[6] = 0;
    g1[7] = 0;
    i32x4 z4 = {0, 0, 0, 0};
#if __clang_major__ >= 23
    i32x8 z8 = {0, 0, 0, 0, 0, 0, 0, 0};
    __builtin_amdgcn_tensor_load_to_lds(g0, g1, z4, z4, z8, 0);
#else
    __builtin_amdgcn_tensor_load_to_lds(g0, g1, z4, z4, 0);
#endif
#if defined(HAVE_TCNT)
    __builtin_amdgcn_s_wait_tensorcnt(0);
#else
    asm volatile("s_wait_tensorcnt 0x0" ::: "memory");
#endif
  }
  // lo plane: per-lane async Global->LDS copies (ASYNCcnt)
  for (int c = threadIdx.x; c < 16 * NLAT / 8; c += blockDim.x)
    async_lds_b128((unsigned)(uintptr_t)&lAlo[c * 8],
                   (unsigned long long)(uintptr_t)&srcl[c * 8]);
#else
  for (int c = threadIdx.x; c < 16 * NLAT / 8; c += blockDim.x) {
    async_lds_b128((unsigned)(uintptr_t)&lAhi[c * 8],
                   (unsigned long long)(uintptr_t)&srch[c * 8]);
    async_lds_b128((unsigned)(uintptr_t)&lAlo[c * 8],
                   (unsigned long long)(uintptr_t)&srcl[c * 8]);
  }
#endif
  asm volatile("s_wait_asynccnt 0x0" ::: "memory");
  __syncthreads();

  const int wave = threadIdx.x >> 5, lane = threadIdx.x & 31;
  const int hl = lane >> 4, ln = lane & 15;

#pragma unroll
  for (int lt = 0; lt < 2; ++lt) {
    const int l0 = (wave * 2 + lt) * 16;
    const bf16_t* Bh = Whi + ((size_t)m * LMAX + (l0 + ln)) * NLAT;  // k-contig
    const bf16_t* Bl = Wlo + ((size_t)m * LMAX + (l0 + ln)) * NLAT;
    __builtin_prefetch(Bh, 0, 3);
    __builtin_prefetch(Bl, 0, 3);
    v8f acc = {};
#pragma unroll
    for (int k0 = 0; k0 < NLAT; k0 += 32) {
      const int ka = k0 + 8 * hl, kb = k0 + 8 * hl;
      v16bf ah = frag2(&lAhi[ln * NLAT + ka], &lAhi[ln * NLAT + ka + 16]);
      v16bf al = frag2(&lAlo[ln * NLAT + ka], &lAlo[ln * NLAT + ka + 16]);
      v16bf bh = frag2(&Bh[kb], &Bh[kb + 16]);
      v16bf bl = frag2(&Bl[kb], &Bl[kb + 16]);
      acc = WMMA_BF16(ah, bh, acc);
      acc = WMMA_BF16(ah, bl, acc);
      acc = WMMA_BF16(al, bh, acc);
    }
#pragma unroll
    for (int i = 0; i < 8; ++i) {
      const int r = r0 + i + 8 * hl, l = l0 + ln;
      out[(((size_t)r * LMAX + l) * MMAX + m) * 2 + part] = acc[i];
    }
  }
}

// ==================================================================== f32
__global__ void sht_init_f(float* __restrict__ ws) {
  int idx = blockIdx.x * blockDim.x + threadIdx.x;
  if (idx >= 2 * MPAD * NLON) return;
  int part = idx / (MPAD * NLON);
  int rem  = idx - part * (MPAD * NLON);
  int m = rem / NLON, n = rem - (rem / NLON) * NLON;
  ws[idx] = (m < MMAX) ? dft_val(part, m, n) : 0.0f;
}

__global__ __launch_bounds__(256)
void sht_dft(const float* __restrict__ x, float* __restrict__ ws) {
  __shared__ float lA[16 * NLON];
  const float* Ft = ws;
  float* Y = ws + YOFF_F32;
  const int p0 = blockIdx.x * 16;
  const float* src = x + (size_t)p0 * NLON;
  for (int i = threadIdx.x; i < 16 * NLON / 4; i += blockDim.x)
    ((f4v*)lA)[i] = ((const f4v*)src)[i];
  __syncthreads();
  const int wave = threadIdx.x >> 5, lane = threadIdx.x & 31;
  const int hl = lane >> 4, ln = lane & 15;
  for (int t = wave; t < 34; t += 8) {
    const int part = (t >= 17) ? 1 : 0;
    const int m0 = (t - part * 17) * 16;
    const float* Brow = Ft + (size_t)part * F_ELEMS + (size_t)(m0 + ln) * NLON;
    v8f acc = {};
#pragma unroll 8
    for (int k0 = 0; k0 < NLON; k0 += 4) {
      const int k = k0 + 2 * hl;
      v2f a = *(const v2f*)&lA[ln * NLON + k];
      v2f b = *(const v2f*)&Brow[k];
      acc = WMMA_F32(a, b, acc);
    }
    const int m = m0 + ln;
    if (m < MMAX) {
      float* dst = Y + (size_t)part * Y_PART + (size_t)m * ROWS + p0 + 8 * hl;
#pragma unroll
      for (int i = 0; i < 8; ++i) dst[i] = acc[i];
    }
  }
}

__global__ __launch_bounds__(256)
void sht_leg(const float* __restrict__ wgt, const float* __restrict__ ws_c,
             float* __restrict__ out) {
  __shared__ float lA[16 * NLAT];
  const float* Y = ws_c + YOFF_F32;
  const int m = blockIdx.x, r0 = blockIdx.y * 16, part = blockIdx.z;
  const float* src = Y + (size_t)part * Y_PART + (size_t)m * ROWS + (size_t)r0 * NLAT;
  for (int i = threadIdx.x; i < 16 * NLAT / 4; i += blockDim.x)
    ((f4v*)lA)[i] = ((const f4v*)src)[i];
  __syncthreads();
  const int wave = threadIdx.x >> 5, lane = threadIdx.x & 31;
  const int hl = lane >> 4, ln = lane & 15;
#pragma unroll
  for (int lt = 0; lt < 2; ++lt) {
    const int l0 = (wave * 2 + lt) * 16;
    const float* Brow = wgt + ((size_t)m * LMAX + (l0 + ln)) * NLAT;
    v8f acc = {};
#pragma unroll 8
    for (int k0 = 0; k0 < NLAT; k0 += 4) {
      const int k = k0 + 2 * hl;
      v2f a = *(const v2f*)&lA[ln * NLAT + k];
      v2f b = *(const v2f*)&Brow[k];
      acc = WMMA_F32(a, b, acc);
    }
#pragma unroll
    for (int i = 0; i < 8; ++i) {
      const int r = r0 + i + 8 * hl, l = l0 + ln;
      out[(((size_t)r * LMAX + l) * MMAX + m) * 2 + part] = acc[i];
    }
  }
}

// ==================================================================== launch
extern "C" void kernel_launch(void* const* d_in, const int* in_sizes, int n_in,
                              void* d_out, int out_size, void* d_ws, size_t ws_size,
                              hipStream_t stream) {
  (void)in_sizes; (void)n_in; (void)out_size;
  const float* x = (const float*)d_in[0];       // (4,64,256,512) f32
  const float* w = (const float*)d_in[1];       // (257,256,256) f32
  float* out = (float*)d_out;                   // (4,64,256,257,2) f32
  const int ftot = 2 * MPAD * NLON;

  if (ws_size >= WS_NEED_BF16) {
    char* wsb = (char*)d_ws;
    sht_init_f_bf16<<<(ftot + 255) / 256, 256, 0, stream>>>(wsb);
    sht_cvt_w<<<(int)((W_ELEMS + 255) / 256), 256, 0, stream>>>(w, wsb);
    sht_dft_bf16<<<ROWS / 16, 256, 0, stream>>>(x, wsb);
    dim3 g2(MMAX, BC / 16, 2);
    sht_leg_bf16<<<g2, 256, 0, stream>>>(wsb, out);
  } else {
    float* ws = (float*)d_ws;                   // needs ~136 MB
    sht_init_f<<<(ftot + 255) / 256, 256, 0, stream>>>(ws);
    sht_dft<<<ROWS / 16, 256, 0, stream>>>(x, ws);
    dim3 g2(MMAX, BC / 16, 2);
    sht_leg<<<g2, 256, 0, stream>>>(w, ws, out);
  }
}
